// Attention4_16939351016269
// MI455X (gfx1250) — matrix operand order repaired and checked by reading, not compiled, not run on hardware
//
#include <hip/hip_runtime.h>
#include <hip/hip_bf16.h>

typedef __bf16 bf16_t;
typedef __attribute__((ext_vector_type(8)))  __bf16 bf16x8;
typedef __attribute__((ext_vector_type(16))) __bf16 bf16x16;
typedef __attribute__((ext_vector_type(8)))  float  f32x8;
typedef __attribute__((ext_vector_type(4)))  unsigned int u32x4;
typedef __attribute__((ext_vector_type(8)))  unsigned int u32x8;

#define BATCH 4
#define CDIM 512
#define LDIM 4096
#define HEADS 8
#define DHEAD 64

__device__ __forceinline__ bf16x16 mk16(bf16x8 lo, bf16x8 hi) {
    bf16x16 r;
#pragma unroll
    for (int i = 0; i < 8; ++i) { r[i] = lo[i]; r[i + 8] = hi[i]; }
    return r;
}

__device__ __forceinline__ f32x8 zero8() {
    f32x8 z = {0.f, 0.f, 0.f, 0.f, 0.f, 0.f, 0.f, 0.f};
    return z;
}

// Issue one TDM 2D tile copy: rows x 64B-per-row from row-major global matrix
// (row stride = ktot elements) into LDS with 16B padding per row (row pitch
// 80B), using the Tensor Data Mover.  D# group0/group1 per CDNA5 ISA §8.
__device__ __forceinline__ void tdm_load_a_tile(const bf16_t* gsrc,
                                                unsigned int lds_byte_off,
                                                unsigned int ktot,
                                                unsigned int mdim) {
    unsigned long long ga = (unsigned long long)gsrc;
    u32x4 g0;
    g0[0] = 1u;                                   // count=1 (user descriptor)
    g0[1] = lds_byte_off;                         // lds_addr
    g0[2] = (unsigned int)ga;                     // global_addr[31:0]
    g0[3] = ((unsigned int)(ga >> 32) & 0x1FFFFFFu) | (2u << 30);  // addr[56:32] | type=2
    u32x8 g1;
    g1[0] = (1u << 16)      // data_size = 1 -> 2 bytes
          | (1u << 20)      // pad_enable
          | (3u << 22)      // pad_interval = 16 DWORDs (one 64B tile row)
          | (3u << 25);     // pad_amount  = 4 DWORDs (16B)
    g1[1] = (ktot & 0xFFFFu) << 16;                               // tensor_dim0 lo
    g1[2] = ((ktot >> 16) & 0xFFFFu) | ((mdim & 0xFFFFu) << 16);  // dim0 hi | dim1 lo
    g1[3] = ((mdim >> 16) & 0xFFFFu) | (32u << 16);               // dim1 hi | tile_dim0=32
    g1[4] = 128u;                                                 // tile_dim1=128 rows
    g1[5] = ktot;                                                 // tensor_dim0_stride lo
    g1[6] = 0u;                                                   // stride hi | dim1_stride lo
    g1[7] = 0u;
    asm volatile("tensor_load_to_lds %0, %1" :: "s"(g0), "s"(g1) : "memory");
}

// ---------------------------------------------------------------- fp32 -> bf16
__global__ __launch_bounds__(256) void cvt_bf16(const float* __restrict__ src,
                                                bf16_t* __restrict__ dst, int n) {
    int i = (blockIdx.x * 256 + threadIdx.x) * 4;
    if (i + 3 < n) {
        float4 f = *(const float4*)(src + i);
        dst[i + 0] = (bf16_t)f.x; dst[i + 1] = (bf16_t)f.y;
        dst[i + 2] = (bf16_t)f.z; dst[i + 3] = (bf16_t)f.w;
    } else {
        for (int j = i; j < n; ++j) dst[j] = (bf16_t)src[j];
    }
}

// w_qdw: [o][i][t] fp32 -> [o][t*512 + i] bf16 (contiguous im2col-K rows)
__global__ __launch_bounds__(256) void prep_qdw(const float* __restrict__ w,
                                                bf16_t* __restrict__ o) {
    int idx = blockIdx.x * 256 + threadIdx.x;
    if (idx < CDIM * CDIM * 3) {
        int oo = idx / (CDIM * 3);
        int rem = idx - oo * (CDIM * 3);
        int ii = rem / 3;
        int t = rem - ii * 3;
        o[oo * (CDIM * 3) + t * CDIM + ii] = (bf16_t)w[idx];
    }
}

// ---------------------------------------------------------------- WMMA GEMM
// Y[M,N] = W[M,Ktot] @ X (per batch). CONV=1: Ktot=3*512 im2col of k=3 pad=1.
// Block tile 128x128, 8 waves, wave tile 32(M)x64(N), K chunk 32.
// A tile staged to LDS by the Tensor Data Mover (padded rows); B tile staged
// transposed by all threads.
template <int CONV, typename OutT>
__global__ __launch_bounds__(256) void gemm_wmma(
    const bf16_t* __restrict__ W, const bf16_t* __restrict__ X,
    OutT* __restrict__ Y, int M, int Ktot, int KI, int N) {
    __shared__ __align__(16) bf16_t at[128 * 40];  // A tile, 80B row pitch (TDM pad)
    __shared__ __align__(16) bf16_t bt[128 * 40];  // B tile transposed [n][k]

    const int lane = threadIdx.x & 31;
    const int wave = threadIdx.x >> 5;
    const int wm = wave & 3;          // M quadrant (x32)
    const int wn = wave >> 2;         // N half (x64)
    const int lg = lane >> 4;         // lane group
    const int ln = lane & 15;
    const int m0 = blockIdx.y * 128;
    const int n0 = blockIdx.x * 128;
    const size_t batch = blockIdx.z;

    const bf16_t* Xb = X + batch * (size_t)KI * N;
    OutT* Yb = Y + batch * (size_t)M * N;
    const unsigned int at_base = (unsigned int)(unsigned long long)(&at[0]);

    f32x8 acc[2][4];
#pragma unroll
    for (int mt = 0; mt < 2; ++mt)
#pragma unroll
        for (int nt = 0; nt < 4; ++nt) acc[mt][nt] = zero8();

    for (int k0 = 0; k0 < Ktot; k0 += 32) {
        __syncthreads();  // previous chunk's fragment reads done

        // ---- TDM: A tile (128 rows x 32 K) -> LDS with padded rows
        if (wave == 0) {
            tdm_load_a_tile(W + (size_t)m0 * Ktot + k0, at_base,
                            (unsigned int)Ktot, (unsigned int)M);
        }

        // ---- stage B tile (32 K-rows x 128 cols) transposed into LDS
#pragma unroll
        for (int it = 0; it < 2; ++it) {
            int t = threadIdx.x + it * 256;      // 0..511
            int r = t >> 4;                      // k row 0..31
            int c8 = (t & 15) << 3;              // col offset 0..120
            int kk = k0 + r;
            if (CONV) {
                int tsh = kk >> 9;               // tap 0..2
                int ich = kk & 511;
                const bf16_t* src = Xb + (size_t)ich * N;
                int colb = n0 + c8 + tsh - 1;
#pragma unroll
                for (int j = 0; j < 8; ++j) {
                    int col = colb + j;
                    bf16_t v = (bf16_t)0.0f;
                    if (col >= 0 && col < N) v = src[col];
                    bt[(c8 + j) * 40 + r] = v;
                }
            } else {
                bf16x8 v = *(const bf16x8*)(Xb + (size_t)kk * N + n0 + c8);
#pragma unroll
                for (int j = 0; j < 8; ++j) bt[(c8 + j) * 40 + r] = v[j];
            }
        }

        if (!CONV && k0 + 32 < Ktot) {  // global_prefetch of next B chunk
            int r = threadIdx.x >> 4;
            int c8 = (threadIdx.x & 15) << 3;
            __builtin_prefetch(Xb + (size_t)(k0 + 32 + r) * N + n0 + c8, 0, 1);
        }

        if (wave == 0) __builtin_amdgcn_s_wait_tensorcnt(0);
        __syncthreads();

        // ---- A fragments from LDS (conflict-free: 80B row pitch)
        bf16x16 af[2];
#pragma unroll
        for (int mt = 0; mt < 2; ++mt) {
            const bf16_t* ar = &at[(wm * 32 + mt * 16 + ln) * 40];
            af[mt] = mk16(*(const bf16x8*)(ar + 8 * lg),
                          *(const bf16x8*)(ar + 16 + 8 * lg));
        }
        // ---- B fragments from LDS + 8 WMMAs
#pragma unroll
        for (int nt = 0; nt < 4; ++nt) {
            const bf16_t* p = &bt[(wn * 64 + nt * 16 + ln) * 40];
            bf16x16 bfv = mk16(*(const bf16x8*)(p + 8 * lg), *(const bf16x8*)(p + 16 + 8 * lg));
#pragma unroll
            for (int mt = 0; mt < 2; ++mt)
                acc[mt][nt] = __builtin_amdgcn_wmma_f32_16x16x32_bf16(
                    false, af[mt], false, bfv, (short)0, acc[mt][nt], false, false);
        }
    }

#pragma unroll
    for (int mt = 0; mt < 2; ++mt)
#pragma unroll
        for (int nt = 0; nt < 4; ++nt) {
            int col = n0 + wn * 64 + nt * 16 + ln;
#pragma unroll
            for (int r = 0; r < 8; ++r) {
                int row = m0 + wm * 32 + mt * 16 + r + 8 * lg;
                Yb[(size_t)row * N + col] = (OutT)acc[mt][nt][r];
            }
        }
}

// ---------------------------------------------------- depthwise k=3 + k/v split
__global__ __launch_bounds__(256) void dwconv_split(
    const bf16_t* __restrict__ kv, const float* __restrict__ wdw,
    bf16_t* __restrict__ kout, bf16_t* __restrict__ vout) {
    int bid = blockIdx.x;            // 0..4095
    int batch = bid >> 10;
    int ch = bid & 1023;
    const bf16_t* src = kv + ((size_t)batch * 1024 + ch) * (size_t)LDIM;
    float w0 = wdw[ch * 3], w1 = wdw[ch * 3 + 1], w2 = wdw[ch * 3 + 2];
    bf16_t* dst = (ch < CDIM)
                      ? kout + ((size_t)batch * CDIM + ch) * (size_t)LDIM
                      : vout + ((size_t)batch * CDIM + (ch - CDIM)) * (size_t)LDIM;
    for (int l = threadIdx.x; l < LDIM; l += 256) {
        float a = (l > 0) ? (float)src[l - 1] : 0.0f;
        float b = (float)src[l];
        float c = (l < LDIM - 1) ? (float)src[l + 1] : 0.0f;
        dst[l] = (bf16_t)(w0 * a + w1 * b + w2 * c);
    }
}

// -------------------------------------------------------- per-row inverse norm
__global__ __launch_bounds__(256) void rownorm(const bf16_t* __restrict__ t,
                                               float* __restrict__ inv, int rowlen) {
    __shared__ float red[256];
    const bf16_t* row = t + (size_t)blockIdx.x * rowlen;
    float s = 0.f;
    for (int l = threadIdx.x; l < rowlen; l += 256) {
        float v = (float)row[l];
        s += v * v;
    }
    red[threadIdx.x] = s;
    __syncthreads();
    for (int off = 128; off > 0; off >>= 1) {
        if (threadIdx.x < off) red[threadIdx.x] += red[threadIdx.x + off];
        __syncthreads();
    }
    if (threadIdx.x == 0)
        inv[blockIdx.x] = 1.0f / fmaxf(sqrtf(red[0]), 1e-12f);
}

// ------------------------------------- fused Gram + softmax + attn@v per (b,h)
__global__ __launch_bounds__(256) void attn_kernel(
    const bf16_t* __restrict__ q, const bf16_t* __restrict__ k,
    const bf16_t* __restrict__ v, const float* __restrict__ invq,
    const float* __restrict__ invk, const float* __restrict__ temp,
    bf16_t* __restrict__ out) {
    __shared__ __align__(16) float gsm[64 * 68];       // raw Gram
    __shared__ __align__(16) bf16_t attn_sm[64 * 72];  // softmaxed attn (bf16)
    __shared__ __align__(16) bf16_t vt[128 * 72];      // v chunk transposed
    __shared__ float iks[64];

    int bh = blockIdx.x;
    int b = bh >> 3, h = bh & 7;
    size_t base = (size_t)b * CDIM + (size_t)h * DHEAD;
    const bf16_t* qb = q + base * LDIM;
    const bf16_t* kb = k + base * LDIM;
    const bf16_t* vb = v + base * LDIM;
    bf16_t* ob = out + base * LDIM;

    int lane = threadIdx.x & 31, wave = threadIdx.x >> 5;
    int lg = lane >> 4, ln = lane & 15;

    if (threadIdx.x < 64) iks[threadIdx.x] = invk[base + threadIdx.x];

    // ---- phase 1: Gram = q @ k^T (64x64, K=4096) via WMMA
    int mt0 = wave >> 2;   // tiles (mt0, nt) and (mt0+2, nt)
    int nt = wave & 3;
    f32x8 acc0 = zero8(), acc1 = zero8();
#pragma unroll 4
    for (int k0 = 0; k0 < LDIM; k0 += 32) {
        const bf16_t* kp = kb + (size_t)(nt * 16 + ln) * LDIM + k0;
        bf16x16 bfv = mk16(*(const bf16x8*)(kp + 8 * lg), *(const bf16x8*)(kp + 16 + 8 * lg));
        const bf16_t* qp0 = qb + (size_t)(mt0 * 16 + ln) * LDIM + k0;
        bf16x16 a0 = mk16(*(const bf16x8*)(qp0 + 8 * lg),
                          *(const bf16x8*)(qp0 + 16 + 8 * lg));
        const bf16_t* qp1 = qb + (size_t)((mt0 + 2) * 16 + ln) * LDIM + k0;
        bf16x16 a1 = mk16(*(const bf16x8*)(qp1 + 8 * lg),
                          *(const bf16x8*)(qp1 + 16 + 8 * lg));
        acc0 = __builtin_amdgcn_wmma_f32_16x16x32_bf16(false, a0, false, bfv,
                                                       (short)0, acc0, false, false);
        acc1 = __builtin_amdgcn_wmma_f32_16x16x32_bf16(false, a1, false, bfv,
                                                       (short)0, acc1, false, false);
    }
#pragma unroll
    for (int r = 0; r < 8; ++r) {
        gsm[(mt0 * 16 + r + 8 * lg) * 68 + nt * 16 + ln] = acc0[r];
        gsm[((mt0 + 2) * 16 + r + 8 * lg) * 68 + nt * 16 + ln] = acc1[r];
    }
    __syncthreads();

    // ---- phase 2: fold norms + temperature, softmax over d (64 cols)
    if (threadIdx.x < 64) {
        int row = threadIdx.x;
        float sc = invq[base + row] * temp[h];
        float mx = -1e30f;
        for (int c = 0; c < 64; ++c)
            mx = fmaxf(mx, gsm[row * 68 + c] * sc * iks[c]);
        float s = 0.f;
        for (int c = 0; c < 64; ++c)
            s += __expf(gsm[row * 68 + c] * sc * iks[c] - mx);
        float is = 1.0f / s;
        for (int c = 0; c < 64; ++c)
            attn_sm[row * 72 + c] =
                (bf16_t)(__expf(gsm[row * 68 + c] * sc * iks[c] - mx) * is);
    }
    __syncthreads();

    // ---- phase 3: out = attn @ v (64 x 4096, K=64) in 128-col chunks
    int mt = wave & 3;
    int ntb = (wave >> 2) * 4;
    for (int nb = 0; nb < LDIM; nb += 128) {
        // stage v chunk transposed
#pragma unroll
        for (int it = 0; it < 4; ++it) {
            int t = threadIdx.x + it * 256;
            int r = t >> 4;            // v row (d) 0..63
            int c8 = (t & 15) << 3;    // chunk col
            bf16x8 vv = *(const bf16x8*)(vb + (size_t)r * LDIM + nb + c8);
#pragma unroll
            for (int j = 0; j < 8; ++j) vt[(c8 + j) * 72 + r] = vv[j];
        }
        __syncthreads();

        f32x8 acc2[4];
#pragma unroll
        for (int j = 0; j < 4; ++j) acc2[j] = zero8();
#pragma unroll
        for (int k0 = 0; k0 < 64; k0 += 32) {
            const bf16_t* ap = &attn_sm[(mt * 16 + ln) * 72 + k0];
            bf16x16 a = mk16(*(const bf16x8*)(ap + 8 * lg),
                             *(const bf16x8*)(ap + 16 + 8 * lg));
#pragma unroll
            for (int j = 0; j < 4; ++j) {
                const bf16_t* bp = &vt[((ntb + j) * 16 + ln) * 72 + k0];
                bf16x16 bv = mk16(*(const bf16x8*)(bp + 8 * lg), *(const bf16x8*)(bp + 16 + 8 * lg));
                acc2[j] = __builtin_amdgcn_wmma_f32_16x16x32_bf16(
                    false, a, false, bv, (short)0, acc2[j], false, false);
            }
        }
#pragma unroll
        for (int j = 0; j < 4; ++j) {
            int col = nb + (ntb + j) * 16 + ln;
#pragma unroll
            for (int r = 0; r < 8; ++r) {
                int row = mt * 16 + r + 8 * lg;
                ob[(size_t)row * LDIM + col] = (bf16_t)acc2[j][r];
            }
        }
        __syncthreads();
    }
}

// ---------------------------------------------------------------- launcher
extern "C" void kernel_launch(void* const* d_in, const int* in_sizes, int n_in,
                              void* d_out, int out_size, void* d_ws, size_t ws_size,
                              hipStream_t stream) {
    const float* x = (const float*)d_in[0];
    const float* y = (const float*)d_in[1];
    const float* w_kv = (const float*)d_in[2];
    const float* w_kv_dw = (const float*)d_in[3];
    const float* w_q = (const float*)d_in[4];
    const float* w_q_dw = (const float*)d_in[5];
    const float* w_proj = (const float*)d_in[6];
    const float* temperature = (const float*)d_in[7];

    const size_t ACT = (size_t)BATCH * CDIM * LDIM * sizeof(bf16_t);  // 16.78 MB
    char* ws = (char*)d_ws;
    bf16_t* R1   = (bf16_t*)(ws);            // x_bf -> y_bf -> out_bf
    bf16_t* R2   = (bf16_t*)(ws + ACT);      // kv_pre (2*ACT) -> q_pre
    bf16_t* R2b  = (bf16_t*)(ws + 2 * ACT);  // q_conv
    bf16_t* KBF  = (bf16_t*)(ws + 3 * ACT);
    bf16_t* VBF  = (bf16_t*)(ws + 4 * ACT);
    char* wp = ws + 5 * ACT;
    bf16_t* WKV  = (bf16_t*)wp;  wp += (size_t)1024 * 512 * 2;
    bf16_t* WQ   = (bf16_t*)wp;  wp += (size_t)512 * 512 * 2;
    bf16_t* WQDW = (bf16_t*)wp;  wp += (size_t)512 * 1536 * 2;
    bf16_t* WPJ  = (bf16_t*)wp;  wp += (size_t)512 * 512 * 2;
    float* INVQ  = (float*)wp;   wp += (size_t)BATCH * CDIM * 4;
    float* INVK  = (float*)wp;

    const int NACT = BATCH * CDIM * LDIM;  // 8,388,608

    // convert activations & weights to bf16
    cvt_bf16<<<(NACT / 4 + 255) / 256, 256, 0, stream>>>(x, R1, NACT);
    cvt_bf16<<<(1024 * 512 / 4 + 255) / 256, 256, 0, stream>>>(w_kv, WKV, 1024 * 512);
    cvt_bf16<<<(512 * 512 / 4 + 255) / 256, 256, 0, stream>>>(w_q, WQ, 512 * 512);
    cvt_bf16<<<(512 * 512 / 4 + 255) / 256, 256, 0, stream>>>(w_proj, WPJ, 512 * 512);
    prep_qdw<<<(512 * 512 * 3 + 255) / 256, 256, 0, stream>>>(w_q_dw, WQDW);

    // kv = W_kv @ x  (1024x512 GEMM)
    gemm_wmma<0, bf16_t><<<dim3(LDIM / 128, 1024 / 128, BATCH), 256, 0, stream>>>(
        WKV, R1, R2, 1024, 512, 512, LDIM);
    // depthwise k=3 + split into k,v
    dwconv_split<<<BATCH * 1024, 256, 0, stream>>>(R2, w_kv_dw, KBF, VBF);
    // q_pre = W_q @ y
    cvt_bf16<<<(NACT / 4 + 255) / 256, 256, 0, stream>>>(y, R1, NACT);
    gemm_wmma<0, bf16_t><<<dim3(LDIM / 128, 512 / 128, BATCH), 256, 0, stream>>>(
        WQ, R1, R2, 512, 512, 512, LDIM);
    // q = dense conv k=3 (im2col GEMM, K=1536)
    gemm_wmma<1, bf16_t><<<dim3(LDIM / 128, 512 / 128, BATCH), 256, 0, stream>>>(
        WQDW, R2, R2b, 512, 1536, 512, LDIM);
    // per-row inverse norms
    rownorm<<<BATCH * CDIM, 256, 0, stream>>>(R2b, INVQ, LDIM);
    rownorm<<<BATCH * CDIM, 256, 0, stream>>>(KBF, INVK, LDIM);
    // fused channel attention
    attn_kernel<<<BATCH * HEADS, 256, 0, stream>>>(R2b, KBF, VBF, INVQ, INVK,
                                                   temperature, R1);
    // final projection -> fp32 output
    gemm_wmma<0, float><<<dim3(LDIM / 128, 512 / 128, BATCH), 256, 0, stream>>>(
        WPJ, R1, (float*)d_out, 512, 512, 512, LDIM);
}
